// JSFSRNet_18751827214901
// MI455X (gfx1250) — matrix operand order repaired and checked by reading, not compiled, not run on hardware
//
#include <hip/hip_runtime.h>
#include <math.h>

typedef __attribute__((ext_vector_type(16))) _Float16 v16h;
typedef __attribute__((ext_vector_type(8)))  float    v8f;
typedef unsigned int u32;
typedef __attribute__((ext_vector_type(4))) u32 v4u;
typedef __attribute__((ext_vector_type(8))) u32 v8u;

#define BN   8
#define HH   160
#define WW   160
#define HWN  (HH*WW)          // 25600
#define NHWT (BN*HWN)         // 204800

__device__ __forceinline__ float apply_act(float v, int act) {
    switch (act) {
        case 1: return v >= 0.0f ? v : 0.05f * v;                          // leaky relu
        case 2: return 0.5f * v * (1.0f + erff(v * 0.7071067811865475f));  // exact gelu
        case 3: return v / (1.0f + expf(-v));                              // silu
        case 4: return fmaxf(v, 0.0f);                                     // relu
        case 5: return 1.0f / (1.0f + expf(-v));                           // sigmoid
        default: return v;
    }
}

// ---------------------------------------------------------------- masks
__global__ void masks_kernel(float* __restrict__ mlow, float* __restrict__ mmid,
                             float* __restrict__ mhigh) {
    int idx = blockIdx.x * blockDim.x + threadIdx.x;
    if (idx >= HWN) return;
    int i = idx / WW, j = idx - (idx / WW) * WW;
    float di = (float)(i - HH / 2), dj = (float)(j - WW / 2);
    float d = sqrtf(di * di + dj * dj);
    if (d == 0.0f) d = 1e-6f;
    float hp = 1.0f / (1.0f + (40.0f / d) * (40.0f / d));
    float lp = 1.0f / (1.0f + (d / 90.0f) * (d / 90.0f));
    mlow[idx]  = 0.8f * lp + 0.5f;
    mmid[idx]  = 0.8f * (hp * lp) + 0.5f;
    mhigh[idx] = 0.8f * hp + 0.5f;
}

// x (B,3,HW) * mask(HW) -> dst slice
__global__ void mask_mul_kernel(const float* __restrict__ x, const float* __restrict__ mask,
                                float* __restrict__ dst, int dstCTot, int dstBase, int total) {
    int idx = blockIdx.x * blockDim.x + threadIdx.x;
    if (idx >= total) return;
    int n = idx / (3 * HWN); int r = idx - n * 3 * HWN; int c = r / HWN; int p = r - c * HWN;
    dst[((size_t)n * dstCTot + dstBase + c) * HWN + p] = x[idx] * mask[p];
}

__global__ void copy_ch_kernel(const float* __restrict__ src, int srcCTot, int srcBase,
                               float* __restrict__ dst, int dstCTot, int dstBase,
                               int C, int total) {
    int idx = blockIdx.x * blockDim.x + threadIdx.x;
    if (idx >= total) return;
    int n = idx / (C * HWN); int r = idx - n * C * HWN; int c = r / HWN; int p = r - c * HWN;
    dst[((size_t)n * dstCTot + dstBase + c) * HWN + p] =
        src[((size_t)n * srcCTot + srcBase + c) * HWN + p];
}

// ---------------------------------------------------------------- weight pack -> f16 [Kpad][CoutPad]
__global__ void pack_w_kernel(const float* __restrict__ w, _Float16* __restrict__ wp,
                              int Cout, int Cin, int KHW, int Kpad, int CoutPad) {
    int idx = blockIdx.x * blockDim.x + threadIdx.x;
    if (idx >= Kpad * CoutPad) return;
    int k = idx / CoutPad, co = idx - (idx / CoutPad) * CoutPad;
    _Float16 v = (_Float16)0.0f;
    if (k < Cin * KHW && co < Cout) {
        int ci = k / KHW, t = k - ci * KHW;
        v = (_Float16)w[((size_t)co * Cin + ci) * KHW + t];
    }
    wp[idx] = v;
}

// ---------------------------------------------------------------- WMMA im2col conv
// 128 threads = 4 wave32s. Block tile: 32 pixels (M) x 64 channels (N).
// Wave w owns N-tile [16w,16w+16); computes two 16x16x(K) GEMM tiles sharing B.
// The whole packed weight slab [Kpad][CoutPad] f16 (<=92KB) is DMAed into LDS
// once per block by the Tensor Data Mover; the K-loop only stages im2col A.
template <int KH>
__global__ __launch_bounds__(128)
void conv_wmma_kernel(const float* __restrict__ in, int CinTot, int ciBase,
                      const _Float16* __restrict__ wp, const float* __restrict__ bias,
                      float* __restrict__ out, int CoutTot, int coBase, int Cout,
                      const float* __restrict__ resid, int residCTot, int residBase,
                      int K, int Kpad, int CoutPad, int act) {
    constexpr int PAD = (KH - 1) / 2;
    constexpr int KHW = KH * KH;
    extern __shared__ _Float16 Bs[];                 // [Kpad][CoutPad] weight slab
    __shared__ _Float16 As[32][33];                  // 32 pixels x 32 K (f16)
    __shared__ unsigned long long pm_b[32];          // per-pixel batch/channel base
    __shared__ int pm_y[32], pm_x[32];

    const int tid  = threadIdx.x;
    const int lane = tid & 31;
    const int wv   = tid >> 5;
    const int tileBase = blockIdx.x * 32;
    const bool active = (wv * 16) < Cout;
    const int co16 = wv * 16;

    // ---- TDM: async DMA of the full weight tensor into LDS (wave 0 issues)
    if (wv == 0) {
        u32 ldsOff = (u32)(size_t)(&Bs[0]);          // low 32 bits of flat addr = LDS offset
        unsigned long long ga = (unsigned long long)(size_t)wp;
        u32 td0 = (u32)CoutPad;                      // tensor dim0 (elements)
        u32 td1 = (u32)Kpad;                         // tensor dim1 (rows)
        unsigned long long s0 = (unsigned long long)CoutPad;               // dim0 stride
        unsigned long long s1 = (unsigned long long)CoutPad * (u32)Kpad;   // dim1 stride
        v4u g0;
        g0[0] = 1u;                                              // count=1 valid user D#
        g0[1] = ldsOff;                                          // lds_addr
        g0[2] = (u32)(ga & 0xffffffffull);                       // global_addr[31:0]
        g0[3] = (u32)((ga >> 32) & 0x01ffffffull) | (2u << 30);  // global_addr[56:32]|type=2
        v8u g1;
        g1[0] = 1u << 16;                                        // data_size=1 (2 bytes)
        g1[1] = (td0 & 0xffffu) << 16;                           // tensor_dim0[15:0]
        g1[2] = (td0 >> 16) | ((td1 & 0xffffu) << 16);           // dim0[31:16]|dim1[15:0]
        g1[3] = (td1 >> 16) | ((u32)CoutPad << 16);              // dim1[31:16]|tile_dim0
        g1[4] = (u32)Kpad;                                       // tile_dim1 (tile_dim2=0)
        g1[5] = (u32)(s0 & 0xffffffffull);                       // dim0_stride[31:0]
        g1[6] = (u32)(s0 >> 32) | (u32)((s1 & 0xffffull) << 16); // |dim1_stride[15:0]
        g1[7] = (u32)(s1 >> 16);                                 // dim1_stride[47:16]
        asm volatile("tensor_load_to_lds %0, %1" :: "s"(g0), "s"(g1) : "memory");
        __builtin_amdgcn_s_wait_tensorcnt(0);
    }
    // ---- per-pixel metadata, once per block
    if (tid < 32) {
        int p = tileBase + tid;
        if (p < NHWT) {
            int n = p / HWN; int rem = p - n * HWN;
            pm_b[tid] = ((unsigned long long)n * CinTot + ciBase) * HWN;
            pm_y[tid] = rem / WW;
            pm_x[tid] = rem - (rem / WW) * WW;
        } else {
            pm_b[tid] = 0; pm_y[tid] = -100000; pm_x[tid] = 0;
        }
    }
    __syncthreads();

    v8f acc0 = {}, acc1 = {};

    for (int kc = 0; kc < Kpad; kc += 32) {
        // -------- stage A: im2col 32x32 patch, fp32 -> f16 (compile-time KHW divisors)
        for (int e = tid; e < 32 * 32; e += 128) {
            int m = e >> 5, kl = e & 31;
            int k = kc + kl;
            _Float16 v = (_Float16)0.0f;
            if (k < K) {
                int ci = k / KHW, t = k - ci * KHW;
                int ky = t / KH,  kx = t - (t / KH) * KH;
                int ys = pm_y[m] + ky - PAD, xs = pm_x[m] + kx - PAD;
                if (ys >= 0 && ys < HH && xs >= 0 && xs < WW)
                    v = (_Float16)in[pm_b[m] + (size_t)ci * HWN + ys * WW + xs];
            }
            As[m][kl] = v;
        }
        __syncthreads();

        if (active) {   // wave-uniform branch: EXEC all-ones inside
            const int g  = lane >> 4;
            const int mr = lane & 15;
            v16h a0, a1, b;
#pragma unroll
            for (int j = 0; j < 16; ++j)   // B 32x16: lane = N, lane group = K half
                b[j] = Bs[(size_t)(kc + g * 16 + j) * CoutPad + co16 + mr];
#pragma unroll
            for (int j = 0; j < 16; ++j) { // A 16x32: lane = M, k = 16*g + j (same order as B)
                int kl = g * 16 + j;
                a0[j] = As[mr][kl];
                a1[j] = As[16 + mr][kl];
            }
            acc0 = __builtin_amdgcn_wmma_f32_16x16x32_f16(false, a0, false, b,
                                                          (short)0, acc0, false, false);
            acc1 = __builtin_amdgcn_wmma_f32_16x16x32_f16(false, a1, false, b,
                                                          (short)0, acc1, false, false);
        }
        __syncthreads();
    }

    if (active) {
        const int g  = lane >> 4;
        const int nc = lane & 15;
        const int co = co16 + nc;
        if (co < Cout) {
            float bsv = bias ? bias[co] : 0.0f;
#pragma unroll
            for (int r = 0; r < 8; ++r) {
                int m = r + 8 * g;          // C/D layout: VGPR r -> M=r / M=r+8
#pragma unroll
                for (int hlf = 0; hlf < 2; ++hlf) {
                    int p = tileBase + hlf * 16 + m;
                    if (p < NHWT) {
                        int n = p / HWN; int rem = p - n * HWN;
                        float v = (hlf ? acc1[r] : acc0[r]) + bsv;
                        v = apply_act(v, act);
                        if (resid)
                            v += resid[((size_t)n * residCTot + residBase + co) * HWN + rem];
                        out[((size_t)n * CoutTot + coBase + co) * HWN + rem] = v;
                    }
                }
            }
        }
    }
}

// ---------------------------------------------------------------- small direct conv (tiny channel counts)
__global__ void conv_small_kernel(const float* __restrict__ in, int CinTot, int ciBase, int Cin,
                                  const float* __restrict__ w, const float* __restrict__ bias,
                                  float* __restrict__ out, int CoutTot, int coBase, int Cout,
                                  int KH, int pad, int act, int total) {
    int idx = blockIdx.x * blockDim.x + threadIdx.x;
    if (idx >= total) return;
    int n = idx / (Cout * HWN); int r = idx - n * Cout * HWN;
    int co = r / HWN; int p = r - co * HWN;
    int y = p / WW, x = p - (p / WW) * WW;
    float acc = bias ? bias[co] : 0.0f;
    for (int ci = 0; ci < Cin; ++ci) {
        const float* ip = in + ((size_t)n * CinTot + ciBase + ci) * HWN;
        for (int ky = 0; ky < KH; ++ky) {
            int ys = y + ky - pad; if (ys < 0 || ys >= HH) continue;
            for (int kx = 0; kx < KH; ++kx) {
                int xs = x + kx - pad; if (xs < 0 || xs >= WW) continue;
                acc += w[(((size_t)co * Cin + ci) * KH + ky) * KH + kx] * ip[ys * WW + xs];
            }
        }
    }
    out[((size_t)n * CoutTot + coBase + co) * HWN + p] = apply_act(acc, act);
}

// ---------------------------------------------------------------- depthwise 3x3 (+bias, +resid)
__global__ void dw3x3_kernel(const float* __restrict__ in, int CinTot, int ciBase,
                             const float* __restrict__ w, const float* __restrict__ bias,
                             const float* __restrict__ resid, int residCTot, int residBase,
                             float* __restrict__ out, int CoutTot, int coBase, int C, int total) {
    int idx = blockIdx.x * blockDim.x + threadIdx.x;
    if (idx >= total) return;
    int n = idx / (C * HWN); int r = idx - n * C * HWN; int c = r / HWN; int p = r - c * HWN;
    int y = p / WW, x = p - (p / WW) * WW;
    const float* ip = in + ((size_t)n * CinTot + ciBase + c) * HWN;
    float acc = bias ? bias[c] : 0.0f;
#pragma unroll
    for (int ky = 0; ky < 3; ++ky) {
        int ys = y + ky - 1; if (ys < 0 || ys >= HH) continue;
#pragma unroll
        for (int kx = 0; kx < 3; ++kx) {
            int xs = x + kx - 1; if (xs < 0 || xs >= WW) continue;
            acc += w[c * 9 + ky * 3 + kx] * ip[ys * WW + xs];
        }
    }
    if (resid) acc += resid[((size_t)n * residCTot + residBase + c) * HWN + p];
    out[((size_t)n * CoutTot + coBase + c) * HWN + p] = acc;
}

// ---------------------------------------------------------------- batchnorm stats over (B,H,W)
__global__ void bn_stats_kernel(const float* __restrict__ x, int CTot, int cBase,
                                float* __restrict__ mean, float* __restrict__ var) {
    __shared__ float s1[256], s2[256];
    int c = blockIdx.x;
    float a = 0.f, b = 0.f;
    for (int i = threadIdx.x; i < NHWT; i += 256) {
        int n = i / HWN, p = i - n * HWN;
        float v = x[((size_t)n * CTot + cBase + c) * HWN + p];
        a += v; b += v * v;
    }
    s1[threadIdx.x] = a; s2[threadIdx.x] = b;
    __syncthreads();
    for (int s = 128; s > 0; s >>= 1) {
        if (threadIdx.x < s) { s1[threadIdx.x] += s1[threadIdx.x + s]; s2[threadIdx.x] += s2[threadIdx.x + s]; }
        __syncthreads();
    }
    if (threadIdx.x == 0) {
        float mu = s1[0] / (float)NHWT;
        mean[c] = mu;
        var[c]  = fmaxf(s2[0] / (float)NHWT - mu * mu, 0.f);
    }
}

__global__ void bn_silu_kernel(const float* __restrict__ x, int CTot, int cBase,
                               const float* __restrict__ mean, const float* __restrict__ var,
                               const float* __restrict__ g, const float* __restrict__ b,
                               float* __restrict__ dst, int dstCTot, int dstBase,
                               int C, int total) {
    int idx = blockIdx.x * blockDim.x + threadIdx.x;
    if (idx >= total) return;
    int n = idx / (C * HWN); int r = idx - n * C * HWN; int c = r / HWN; int p = r - c * HWN;
    float v = x[((size_t)n * CTot + cBase + c) * HWN + p];
    v = (v - mean[c]) * rsqrtf(var[c] + 1e-5f) * g[c] + b[c];
    v = v / (1.0f + expf(-v));
    dst[((size_t)n * dstCTot + dstBase + c) * HWN + p] = v;
}

// ---------------------------------------------------------------- CCA
__global__ void cca_stats_kernel(const float* __restrict__ x, int C, float* __restrict__ y) {
    __shared__ float s1[256], s2[256];
    int n = blockIdx.x / C, c = blockIdx.x - (blockIdx.x / C) * C;
    const float* ip = x + ((size_t)n * C + c) * HWN;
    float a = 0.f, b = 0.f;
    for (int i = threadIdx.x; i < HWN; i += 256) { float v = ip[i]; a += v; b += v * v; }
    s1[threadIdx.x] = a; s2[threadIdx.x] = b;
    __syncthreads();
    for (int s = 128; s > 0; s >>= 1) {
        if (threadIdx.x < s) { s1[threadIdx.x] += s1[threadIdx.x + s]; s2[threadIdx.x] += s2[threadIdx.x + s]; }
        __syncthreads();
    }
    if (threadIdx.x == 0) {
        float mu = s1[0] / (float)HWN;
        float vr = fmaxf(s2[0] / (float)HWN - mu * mu, 0.f);
        y[blockIdx.x] = sqrtf(vr) + mu;
    }
}

__global__ void cca_mlp1_kernel(const float* __restrict__ y, const float* __restrict__ w,
                                const float* __restrict__ b, float* __restrict__ z) {
    int idx = blockIdx.x * blockDim.x + threadIdx.x;
    if (idx >= BN * 3) return;
    int n = idx / 3, co = idx - (idx / 3) * 3;
    float s = b[co];
    for (int ci = 0; ci < 56; ++ci) s += w[co * 56 + ci] * y[n * 56 + ci];
    z[idx] = fmaxf(s, 0.0f);
}

__global__ void cca_mlp2_kernel(const float* __restrict__ z, const float* __restrict__ w,
                                const float* __restrict__ b, float* __restrict__ sc) {
    int idx = blockIdx.x * blockDim.x + threadIdx.x;
    if (idx >= BN * 56) return;
    int n = idx / 56, co = idx - (idx / 56) * 56;
    float s = b[co];
    for (int ci = 0; ci < 3; ++ci) s += w[co * 3 + ci] * z[n * 3 + ci];
    sc[idx] = 1.0f / (1.0f + expf(-s));
}

__global__ void cca_scale_kernel(const float* __restrict__ in, const float* __restrict__ sc,
                                 float* __restrict__ out, int total) {
    int idx = blockIdx.x * blockDim.x + threadIdx.x;
    if (idx >= total) return;
    int n = idx / (56 * HWN); int r = idx - n * 56 * HWN; int c = r / HWN;
    out[idx] = in[idx] * sc[n * 56 + c];
}

// ---------------------------------------------------------------- pixel shuffle r=3
__global__ void pixel_shuffle_kernel(const float* __restrict__ in, float* __restrict__ out) {
    const int OH = HH * 3, OW = WW * 3;
    int idx = blockIdx.x * blockDim.x + threadIdx.x;
    int total = BN * 3 * OH * OW;
    if (idx >= total) return;
    int n = idx / (3 * OH * OW); int r = idx - n * 3 * OH * OW;
    int co = r / (OH * OW); int rr = r - co * OH * OW;
    int oy = rr / OW; int ox = rr - oy * OW;
    int iy = oy / 3, ry = oy - iy * 3;
    int ix = ox / 3, rx = ox - ix * 3;
    int c = co * 9 + ry * 3 + rx;
    out[idx] = in[((size_t)n * 27 + c) * HWN + iy * WW + ix];
}

// ================================================================ host
extern "C" void kernel_launch(void* const* d_in, const int* in_sizes, int n_in,
                              void* d_out, int out_size, void* d_ws, size_t ws_size,
                              hipStream_t stream) {
    (void)in_sizes; (void)n_in; (void)out_size; (void)ws_size;
    // ---- input pointers (setup_inputs() insertion order, nested dicts depth-first)
    int ii = 0;
    const float* X = (const float*)d_in[ii++];
    const float *fdb_pw[3], *fdb_dw[3], *fdb_dwb[3], *fdb_g[3], *fdb_b[3];
    for (int b = 0; b < 3; ++b) {  // low, mid, high
        fdb_pw[b]  = (const float*)d_in[ii++];
        fdb_dw[b]  = (const float*)d_in[ii++];
        fdb_dwb[b] = (const float*)d_in[ii++];
        fdb_g[b]   = (const float*)d_in[ii++];
        fdb_b[b]   = (const float*)d_in[ii++];
    }
    const float* fea1_pw  = (const float*)d_in[ii++];
    const float* fea1_dw  = (const float*)d_in[ii++];
    const float* fea1_dwb = (const float*)d_in[ii++];
    const float* fea2_pw  = (const float*)d_in[ii++];
    const float* fea2_dw  = (const float*)d_in[ii++];
    const float* fea2_dwb = (const float*)d_in[ii++];
    const float *c1w[10], *c1b[10], *c2w[10], *c2b[10], *c3w[10], *c3b[10],
                *c4w[10], *c4b[10], *c5w[10], *c5b[10],
                *aw1[10], *ab1[10], *aw2[10], *ab2[10];
    for (int i = 0; i < 10; ++i) {
        c1w[i] = (const float*)d_in[ii++]; c1b[i] = (const float*)d_in[ii++];
        c2w[i] = (const float*)d_in[ii++]; c2b[i] = (const float*)d_in[ii++];
        c3w[i] = (const float*)d_in[ii++]; c3b[i] = (const float*)d_in[ii++];
        c4w[i] = (const float*)d_in[ii++]; c4b[i] = (const float*)d_in[ii++];
        c5w[i] = (const float*)d_in[ii++]; c5b[i] = (const float*)d_in[ii++];
        aw1[i] = (const float*)d_in[ii++]; ab1[i] = (const float*)d_in[ii++];
        aw2[i] = (const float*)d_in[ii++]; ab2[i] = (const float*)d_in[ii++];
    }
    const float* tc1w  = (const float*)d_in[ii++];
    const float* tc1b  = (const float*)d_in[ii++];
    const float* c2pw  = (const float*)d_in[ii++];
    const float* c2dw  = (const float*)d_in[ii++];
    const float* c2dwb = (const float*)d_in[ii++];
    const float* upw   = (const float*)d_in[ii++];
    const float* upb   = (const float*)d_in[ii++];

    // ---- workspace bump allocator
    char* wsb = (char*)d_ws; size_t off = 0;
    auto alloc = [&](size_t bytes) -> void* {
        void* r = wsb + off; off = (off + bytes + 255) & ~(size_t)255; return r;
    };
    const size_t CH = (size_t)NHWT * sizeof(float);  // one fp32 channel-map across batch
    float* masks  = (float*)alloc(3 * HWN * sizeof(float));
    float* fd[3]; for (int b = 0; b < 3; ++b) fd[b] = (float*)alloc(3 * CH);
    float* yb     = (float*)alloc(3 * CH);
    float* bnm    = (float*)alloc(64 * sizeof(float));
    float* bnv    = (float*)alloc(64 * sizeof(float));
    float* fdbcat = (float*)alloc(6 * CH);
    float* tmp56  = (float*)alloc(56 * CH);
    float* fea_in = (float*)alloc(12 * CH);
    float* ofea   = (float*)alloc(56 * CH);
    float* o1     = (float*)alloc(56 * CH);
    float* o2     = (float*)alloc(56 * CH);
    float* o3     = (float*)alloc(56 * CH);
    float* catb   = (float*)alloc(56 * CH);
    float* ccaout = (float*)alloc(56 * CH);
    float* ccay   = (float*)alloc(BN * 56 * sizeof(float));
    float* ccaz   = (float*)alloc(BN * 3 * sizeof(float));
    float* ccas   = (float*)alloc(BN * 56 * sizeof(float));
    float* trunk  = (float*)alloc((size_t)728 * CH);
    float* y1     = (float*)alloc(56 * CH);
    float* y2     = (float*)alloc(56 * CH);
    float* up27   = (float*)alloc(27 * CH);
    _Float16* wp  = (_Float16*)alloc((size_t)736 * 64 * sizeof(_Float16)); // max packed weight

    const float* bandmask[3] = {masks, masks + HWN, masks + 2 * HWN};

    auto grid1d = [](int total) { return dim3((total + 255) / 256); };

    // generic WMMA conv: pack weights f16, then TDM+WMMA GEMM-conv
    auto conv_wmma = [&](const float* in, int CinTot, int ciBase, int Cin,
                         const float* w, const float* bias,
                         float* out, int CoutTot, int coBase, int Cout,
                         const float* resid, int residCTot, int residBase,
                         int KH, int act) {
        int KHW = KH * KH, K = Cin * KHW;
        int Kpad = (K + 31) & ~31, CoutPad = (Cout + 15) & ~15;
        int np = Kpad * CoutPad;
        pack_w_kernel<<<grid1d(np), 256, 0, stream>>>(w, wp, Cout, Cin, KHW, Kpad, CoutPad);
        size_t shbytes = (size_t)Kpad * CoutPad * sizeof(_Float16);
        dim3 g((NHWT + 31) / 32);
        if (KH == 1)
            conv_wmma_kernel<1><<<g, 128, shbytes, stream>>>(
                in, CinTot, ciBase, wp, bias, out, CoutTot, coBase, Cout,
                resid, residCTot, residBase, K, Kpad, CoutPad, act);
        else
            conv_wmma_kernel<3><<<g, 128, shbytes, stream>>>(
                in, CinTot, ciBase, wp, bias, out, CoutTot, coBase, Cout,
                resid, residCTot, residBase, K, Kpad, CoutPad, act);
    };

    // ---- masks
    masks_kernel<<<grid1d(HWN), 256, 0, stream>>>(masks, masks + HWN, masks + 2 * HWN);

    // ---- FDB branches + freq features (trunk channels 560 + 56*band)
    for (int b = 0; b < 3; ++b) {
        int t3 = BN * 3 * HWN;
        mask_mul_kernel<<<grid1d(t3), 256, 0, stream>>>(X, bandmask[b], fd[b], 3, 0, t3);
        // bsconv: pw 3->3 (1x1, no bias) then dw 3x3 (+bias)
        conv_small_kernel<<<grid1d(t3), 256, 0, stream>>>(fd[b], 3, 0, 3, fdb_pw[b], nullptr,
                                                         tmp56, 3, 0, 3, 1, 0, 0, t3);
        dw3x3_kernel<<<grid1d(t3), 256, 0, stream>>>(tmp56, 3, 0, fdb_dw[b], fdb_dwb[b],
                                                     nullptr, 0, 0, yb, 3, 0, 3, t3);
        bn_stats_kernel<<<dim3(3), 256, 0, stream>>>(yb, 3, 0, bnm, bnv);
        bn_silu_kernel<<<grid1d(t3), 256, 0, stream>>>(yb, 3, 0, bnm, bnv, fdb_g[b], fdb_b[b],
                                                       fdbcat, 6, 0, 3, t3);
        copy_ch_kernel<<<grid1d(t3), 256, 0, stream>>>(fd[b], 3, 0, fdbcat, 6, 3, 3, t3);
        // fea2 bsconv on fdbcat -> trunk slice
        conv_wmma(fdbcat, 6, 0, 6, fea2_pw, nullptr, tmp56, 56, 0, 56, nullptr, 0, 0, 1, 0);
        int t56 = BN * 56 * HWN;
        dw3x3_kernel<<<grid1d(t56), 256, 0, stream>>>(tmp56, 56, 0, fea2_dw, fea2_dwb,
                                                      nullptr, 0, 0, trunk, 728, 560 + b * 56, 56, t56);
    }

    // ---- fea1: concat [x, x*m_low, x*m_mid, x*m_high] then bsconv 12->56
    {
        int t3 = BN * 3 * HWN;
        copy_ch_kernel<<<grid1d(t3), 256, 0, stream>>>(X, 3, 0, fea_in, 12, 0, 3, t3);
        for (int b = 0; b < 3; ++b)
            copy_ch_kernel<<<grid1d(t3), 256, 0, stream>>>(fd[b], 3, 0, fea_in, 12, 3 + 3 * b, 3, t3);
        conv_wmma(fea_in, 12, 0, 12, fea1_pw, nullptr, tmp56, 56, 0, 56, nullptr, 0, 0, 1, 0);
        int t56 = BN * 56 * HWN;
        dw3x3_kernel<<<grid1d(t56), 256, 0, stream>>>(tmp56, 56, 0, fea1_dw, fea1_dwb,
                                                      nullptr, 0, 0, ofea, 56, 0, 56, t56);
    }

    // ---- 10 x SDB; block i writes trunk slice i, residual from its input
    int t56 = BN * 56 * HWN, t14 = BN * 14 * HWN;
    for (int i = 0; i < 10; ++i) {
        const float* hin; int hCT, hCB;
        if (i == 0) { hin = ofea;  hCT = 56;  hCB = 0; }
        else        { hin = trunk; hCT = 728; hCB = (i - 1) * 56; }

        conv_wmma(hin, hCT, hCB, 56, c1w[i], c1b[i], o1, 56, 0, 56, nullptr, 0, 0, 3, 1);
        copy_ch_kernel<<<grid1d(t14), 256, 0, stream>>>(o1, 56, 0, catb, 56, 0, 14, t14);
        conv_wmma(o1, 56, 14, 42, c2w[i], c2b[i], o2, 56, 0, 56, nullptr, 0, 0, 3, 1);
        copy_ch_kernel<<<grid1d(t14), 256, 0, stream>>>(o2, 56, 0, catb, 56, 14, 14, t14);
        conv_wmma(o2, 56, 14, 42, c3w[i], c3b[i], o3, 56, 0, 56, nullptr, 0, 0, 3, 1);
        copy_ch_kernel<<<grid1d(t14), 256, 0, stream>>>(o3, 56, 0, catb, 56, 28, 14, t14);
        conv_wmma(o3, 56, 14, 42, c4w[i], c4b[i], catb, 56, 42, 14, nullptr, 0, 0, 3, 0);
        // CCA
        cca_stats_kernel<<<dim3(BN * 56), 256, 0, stream>>>(catb, 56, ccay);
        cca_mlp1_kernel<<<dim3(1), 32, 0, stream>>>(ccay, aw1[i], ab1[i], ccaz);
        cca_mlp2_kernel<<<grid1d(BN * 56), 256, 0, stream>>>(ccaz, aw2[i], ab2[i], ccas);
        cca_scale_kernel<<<grid1d(t56), 256, 0, stream>>>(catb, ccas, ccaout, t56);
        // c5 1x1 + residual, straight into trunk slice i
        conv_wmma(ccaout, 56, 0, 56, c5w[i], c5b[i], trunk, 728, i * 56, 56, hin, hCT, hCB, 1, 0);
    }

    // ---- trunk 1x1 (728->56) + exact GELU
    conv_wmma(trunk, 728, 0, 728, tc1w, tc1b, y1, 56, 0, 56, nullptr, 0, 0, 1, 2);
    // ---- c2 bsconv + out_fea residual
    conv_wmma(y1, 56, 0, 56, c2pw, nullptr, tmp56, 56, 0, 56, nullptr, 0, 0, 1, 0);
    dw3x3_kernel<<<grid1d(t56), 256, 0, stream>>>(tmp56, 56, 0, c2dw, c2dwb,
                                                  ofea, 56, 0, y2, 56, 0, 56, t56);
    // ---- up conv 56->27 (3x3) then pixel-shuffle x3
    conv_wmma(y2, 56, 0, 56, upw, upb, up27, 27, 0, 27, nullptr, 0, 0, 3, 0);
    pixel_shuffle_kernel<<<grid1d(BN * 3 * (HH * 3) * (WW * 3)), 256, 0, stream>>>(up27, (float*)d_out);
}
